// CorvidBrain_15195594293721
// MI455X (gfx1250) — matrix operand order repaired and checked by reading, not compiled, not run on hardware
//
#include <hip/hip_runtime.h>

// ---------------------------------------------------------------------------
// CorvidBrain fused RNN for MI455X (gfx1250, wave32, WMMA bf16 16x16x32).
// One wave = one 16-row batch tile; full T=256 scan inside the kernel.
// All matmuls via v_wmma_f32_16x16x32_bf16 (bf16 inputs, f32 accumulate).
// Weights pre-packed into WMMA B-fragment order in d_ws by pack_weight().
// ---------------------------------------------------------------------------

typedef __attribute__((ext_vector_type(16))) __bf16 v16bf;
typedef __attribute__((ext_vector_type(8)))  __bf16 v8bf;
typedef __attribute__((ext_vector_type(8)))  float  v8f;

#define XK 128   // staging-buffer row stride (bf16 elems), max K tiles = 4
#define SP 24    // state row stride (f32), max h = 23

// packed-weight offsets in bf16 elements (each 16x32 tile = 512 elems)
#define PO_INPUT  0
#define PO_WULST  512
#define PO_TECTUM 1024
#define PO_MESO   1536
#define PO_HIPPO  3584
#define PO_NCL    5632
#define PO_SONG   8704
#define PO_CER    10240
#define PO_ARCO   11264
#define PO_H2W_G  12288
#define PO_H2W_P  12800
#define PO_N2M_G  13312
#define PO_N2M_P  14336
#define PO_S2A_G  15360
#define PO_S2A_P  16896
#define PO_TGATE  18432
#define PO_TPROJ  34816
#define PO_NEXT   51200
#define PO_PROP1  53248
// end = 61440 bf16 = 122880 bytes of workspace

#define CB_B 1024
#define CB_T 256
#define CB_H 128

struct BrainArgs {
    const float* in[83];
    const __bf16* wp;
    float* hid;
};

// ---------------------------------------------------------------------------
// Pack W[n][k] (row-major f32, K = KA + KB, columns k<KA from A, else Bm)
// into WMMA B-fragment order: tile(nt,kt), lane L, elem e:
//   n = nt*16 + (L&15),  k = kt*32 + 8*(L>>4) + (e&7) + 16*(e>>3)
// ---------------------------------------------------------------------------
__global__ void pack_weight(const float* __restrict__ A, const float* __restrict__ Bm,
                            int N, int KA, int KB, int nT, int kT,
                            __bf16* __restrict__ dst)
{
    int total = nT * kT * 512;
    int K = KA + KB;
    for (int i = blockIdx.x * blockDim.x + threadIdx.x; i < total;
         i += gridDim.x * blockDim.x) {
        int tile = i >> 9;
        int r    = i & 511;
        int lane = r >> 4;
        int e    = r & 15;
        int nt = tile / kT, kt = tile % kT;
        int n = nt * 16 + (lane & 15);
        int k = kt * 32 + ((lane >> 4) << 3) + (e & 7) + ((e >> 3) << 4);
        float val = 0.0f;
        if (n < N && k < K)
            val = (k < KA) ? A[(size_t)n * KA + k] : Bm[(size_t)n * KB + (k - KA)];
        dst[(size_t)tile * 512 + lane * 16 + e] = (__bf16)val;
    }
}

// ---------------------------------------------------------------------------
// Y[16][N] = Xbf16[16][kT*32] @ W^T + bias, via WMMA 16x16x32 bf16.
// xb: LDS bf16, row stride XK, zero-padded to kT*32 columns.
// wp: packed fragments [nt][kt][32 lanes][16 bf16].
// ---------------------------------------------------------------------------
__device__ inline void gemm16(const __bf16* __restrict__ wp, int nT, int kT,
                              const __bf16* xb, const float* __restrict__ bias,
                              int N, float* yb, int ldy, int lane)
{
    const int half = lane >> 4, lm = lane & 15;
    for (int nt = 0; nt < nT; ++nt) {
        int n = nt * 16 + lm;
        float bv = (bias != nullptr && n < N) ? bias[n] : 0.0f;
        v8f acc;
#pragma unroll
        for (int r = 0; r < 8; ++r) acc[r] = bv;
        for (int kt = 0; kt < kT; ++kt) {
            // A fragment (ISA 16-bit A layout): lane holds row m=lm,
            // elems 0..7  -> k = kt*32 + 8*half + 0..7
            // elems 8..15 -> k = kt*32 + 8*half + 16..23
            const __bf16* xr = xb + lm * XK + kt * 32 + half * 8;
            v8bf alo = *(const v8bf*)xr;
            v8bf ahi = *(const v8bf*)(xr + 16);
            v16bf a;
#pragma unroll
            for (int i = 0; i < 8; ++i) { a[i] = alo[i]; a[8 + i] = ahi[i]; }
            v16bf b = *(const v16bf*)(wp + (size_t)(nt * kT + kt) * 512 + lane * 16);
            acc = __builtin_amdgcn_wmma_f32_16x16x32_bf16(
                false, a, false, b, (short)0, acc, false, false);
        }
#pragma unroll
        for (int r = 0; r < 8; ++r) {
            int m = r + half * 8;                 // C/D layout: vgpr r, half -> row
            if (n < N) yb[(size_t)m * ldy + n] = acc[r];
        }
    }
}

__device__ inline void xb_zero(__bf16* xb, int kT, int lane)
{
    int cols = kT * 32;
    for (int i = lane; i < 16 * cols; i += 32) {
        int m = i / cols, c = i - m * cols;
        xb[m * XK + c] = (__bf16)0.0f;
    }
}

__device__ inline void xb_put(__bf16* xb, const float* src, int srcLd,
                              int cols, int cofs, int lane)
{
    for (int i = lane; i < 16 * cols; i += 32) {
        int m = i / cols, c = i - m * cols;
        xb[m * XK + cofs + c] = (__bf16)src[m * srcLd + c];
    }
}

__device__ inline float sigm(float x) { return 1.0f / (1.0f + expf(-x)); }

// axon: dst[16][dout] += coef * sigmoid(x@gW^T+gb) * tanh(x@pW^T+pb), x = src (LDS state)
__device__ inline void axon_add(const __bf16* wpG, const __bf16* wpP, int nT,
                                const float* biasG, const float* biasP, int dout,
                                const float* src, int scols, float coef,
                                float* dst, __bf16* xb, float* yb, float* tB, int lane)
{
    xb_zero(xb, 1, lane);
    xb_put(xb, src, SP, scols, 0, lane);
    __syncthreads();
    gemm16(wpG, nT, 1, xb, biasG, dout, yb, 128, lane);
    gemm16(wpP, nT, 1, xb, biasP, dout, tB, 128, lane);
    __syncthreads();
    for (int i = lane; i < 16 * dout; i += 32) {
        int m = i / dout, j = i - m * dout;
        dst[m * 128 + j] += coef * sigm(yb[m * 128 + j]) * tanhf(tB[m * 128 + j]);
    }
    __syncthreads();
}

// liquid-cell step + layernorm. Input = [e0 | e1? | (S if self) | S] -> [Wi|Wh].
__device__ inline void nuc_step(const __bf16* wp, int nT, int kT,
                                const float* e0, int ld0, int c0,
                                const float* e1, int ld1, int c1,
                                int self, float* St, int h,
                                const float* bias, const float* ls, float dts,
                                const float* g, const float* bln,
                                __bf16* xb, float* yb, int lane)
{
    xb_zero(xb, kT, lane);
    int c = 0;
    xb_put(xb, e0, ld0, c0, 0, lane); c = c0;
    if (e1) { xb_put(xb, e1, ld1, c1, c, lane); c += c1; }
    if (self) { xb_put(xb, St, SP, h, c, lane); c += h; }
    xb_put(xb, St, SP, h, c, lane);          // Wh operand
    __syncthreads();
    gemm16(wp, nT, kT, xb, bias, h, yb, 128, lane);
    __syncthreads();
    for (int i = lane; i < 16 * h; i += 32) {
        int m = i / h, j = i - m * h;
        float hv = St[m * SP + j];
        float stp = dts * expf(ls[j]);
        float hn = hv + stp * (tanhf(yb[m * 128 + j]) - hv);
        yb[m * 128 + j] = hn;
    }
    __syncthreads();
    if (lane < 16) {
        int m = lane;
        float mean = 0.0f;
        for (int j = 0; j < h; ++j) mean += yb[m * 128 + j];
        mean /= (float)h;
        float var = 0.0f;
        for (int j = 0; j < h; ++j) { float d = yb[m * 128 + j] - mean; var += d * d; }
        var /= (float)h;
        float inv = rsqrtf(var + 1e-5f);
        for (int j = 0; j < h; ++j)
            St[m * SP + j] = (yb[m * 128 + j] - mean) * inv * g[j] + bln[j];
    }
    __syncthreads();
}

// ---------------------------------------------------------------------------
// Main scan kernel: 1 wave per block, 16 batch rows per wave, 256 steps.
// ---------------------------------------------------------------------------
__global__ __launch_bounds__(32) void corvid_rnn(BrainArgs A)
{
    __shared__ float S[8][16 * SP];                 // region states (f32)
    __shared__ alignas(16) __bf16 xb[16 * XK];      // bf16 A-operand staging
    __shared__ float yb[16 * 128];
    __shared__ float tA[16 * 128];
    __shared__ float tB[16 * 128];

    const int lane = threadIdx.x;
    const int b0 = blockIdx.x * 16;
    const float* x = A.in[0];
    const __bf16* W = A.wp;

    for (int r = 0; r < 8; ++r)
        for (int i = lane; i < 16 * SP; i += 32) S[r][i] = 0.0f;
    __syncthreads();

    for (int t = 0; t < CB_T; ++t) {
        if (t + 1 < CB_T)
            __builtin_prefetch(x + ((size_t)(b0 + (lane & 15)) * CB_T + t + 1) * 16, 0, 1);

        // 1. inp = xt @ input_proj -> tA[16][15]
        xb_zero(xb, 1, lane);
        for (int i = lane; i < 16 * 16; i += 32) {
            int m = i >> 4, k = i & 15;
            xb[m * XK + k] = (__bf16)x[((size_t)(b0 + m) * CB_T + t) * 16 + k];
        }
        __syncthreads();
        gemm16(W + PO_INPUT, 1, 1, xb, A.in[2], 15, tA, 128, lane);
        __syncthreads();

        // 2. w_in += 0.1*mask * axon(hippo_to_wulst, hippo_old)
        if (t > 0)
            axon_add(W + PO_H2W_G, W + PO_H2W_P, 1, A.in[54], A.in[52], 15,
                     S[3], 19, 0.1f, tA, xb, yb, tB, lane);

        // 3. wulst (no self): [w_in(15), h(15)]
        nuc_step(W + PO_WULST, 1, 1, tA, 128, 15, nullptr, 0, 0, 0, S[0], 15,
                 A.in[4], A.in[6], 2.0f, A.in[7], A.in[8], xb, yb, lane);
        // 4. tectum (no self): [wulst(15), h(10)]
        nuc_step(W + PO_TECTUM, 1, 1, S[0], SP, 15, nullptr, 0, 0, 0, S[1], 10,
                 A.in[10], A.in[12], 2.0f, A.in[13], A.in[14], xb, yb, lane);

        // 5. meso_in = [wulst, tectum] + 0.05*mask * axon(ncl_to_meso, ncl_old)
        for (int i = lane; i < 16 * 15; i += 32) {
            int m = i / 15, j = i - m * 15; tA[m * 128 + j] = S[0][m * SP + j];
        }
        for (int i = lane; i < 16 * 10; i += 32) {
            int m = i / 10, j = i - m * 10; tA[m * 128 + 15 + j] = S[1][m * SP + j];
        }
        __syncthreads();
        if (t > 0)
            axon_add(W + PO_N2M_G, W + PO_N2M_P, 2, A.in[58], A.in[56], 25,
                     S[4], 23, 0.05f, tA, xb, yb, tB, lane);
        nuc_step(W + PO_MESO, 2, 2, tA, 128, 25, nullptr, 0, 0, 1, S[2], 19,
                 A.in[16], A.in[18], 1.0f, A.in[19], A.in[20], xb, yb, lane);

        // 6. hippo (self): [meso(19), h, h]
        nuc_step(W + PO_HIPPO, 2, 2, S[2], SP, 19, nullptr, 0, 0, 1, S[3], 19,
                 A.in[22], A.in[24], 0.6f, A.in[25], A.in[26], xb, yb, lane);
        // 7. ncl (self): [meso(19), hippo(19), h, h]
        nuc_step(W + PO_NCL, 2, 3, S[2], SP, 19, S[3], SP, 19, 1, S[4], 23,
                 A.in[28], A.in[30], 0.4f, A.in[31], A.in[32], xb, yb, lane);
        // 8. song (self): [ncl(23), meso(19), h, h]
        nuc_step(W + PO_SONG, 1, 3, S[4], SP, 23, S[2], SP, 19, 1, S[5], 12,
                 A.in[34], A.in[36], 1.2f, A.in[37], A.in[38], xb, yb, lane);
        // 9. cerebellum (no self): [ncl(23), wulst(15), h]
        nuc_step(W + PO_CER, 1, 2, S[4], SP, 23, S[0], SP, 15, 0, S[6], 15,
                 A.in[40], A.in[42], 1.8f, A.in[43], A.in[44], xb, yb, lane);

        // 10. arco_in = [ncl, cer] + 0.05*mask * axon(song_to_arco, song_new)
        for (int i = lane; i < 16 * 23; i += 32) {
            int m = i / 23, j = i - m * 23; tA[m * 128 + j] = S[4][m * SP + j];
        }
        for (int i = lane; i < 16 * 15; i += 32) {
            int m = i / 15, j = i - m * 15; tA[m * 128 + 23 + j] = S[6][m * SP + j];
        }
        __syncthreads();
        if (t > 0)
            axon_add(W + PO_S2A_G, W + PO_S2A_P, 3, A.in[62], A.in[60], 38,
                     S[5], 12, 0.05f, tA, xb, yb, tB, lane);
        nuc_step(W + PO_ARCO, 1, 2, tA, 128, 38, nullptr, 0, 0, 0, S[7], 12,
                 A.in[46], A.in[48], 1.5f, A.in[49], A.in[50], xb, yb, lane);

        // 11. thalamus: all_r(125) -> gate(128) LN sigmoid * proj(128)
        xb_zero(xb, 4, lane);
        {
            const int hs[8] = {15, 10, 19, 19, 23, 12, 15, 12};
            int ofs = 0;
            for (int r = 0; r < 8; ++r) { xb_put(xb, S[r], SP, hs[r], ofs, lane); ofs += hs[r]; }
        }
        __syncthreads();
        gemm16(W + PO_TGATE, 8, 4, xb, A.in[64], 128, yb, 128, lane);
        gemm16(W + PO_TPROJ, 8, 4, xb, A.in[68], 128, tA, 128, lane);
        __syncthreads();
        if (lane < 16) {
            int m = lane;
            float mean = 0.0f;
            for (int j = 0; j < 128; ++j) mean += yb[m * 128 + j];
            mean *= (1.0f / 128.0f);
            float var = 0.0f;
            for (int j = 0; j < 128; ++j) { float d = yb[m * 128 + j] - mean; var += d * d; }
            var *= (1.0f / 128.0f);
            float inv = rsqrtf(var + 1e-5f);
            for (int j = 0; j < 128; ++j) {
                float v = (yb[m * 128 + j] - mean) * inv * A.in[65][j] + A.in[66][j];
                yb[m * 128 + j] = sigm(v);
            }
        }
        __syncthreads();
        for (int i = lane; i < 16 * 128; i += 32) {
            int m = i >> 7, j = i & 127;
            A.hid[((size_t)(b0 + m) * CB_T + t) * CB_H + j] = yb[m * 128 + j] * tA[m * 128 + j];
        }
        __syncthreads();
    }
}

// ---------------------------------------------------------------------------
// Post-processing
// ---------------------------------------------------------------------------
__device__ inline void row_ln_to_bf16(const float* row, int L, const float* g,
                                      const float* b, __bf16* dst)
{
    float mean = 0.0f;
    for (int j = 0; j < L; ++j) mean += row[j];
    mean /= (float)L;
    float var = 0.0f;
    for (int j = 0; j < L; ++j) { float d = row[j] - mean; var += d * d; }
    var /= (float)L;
    float inv = rsqrtf(var + 1e-5f);
    for (int j = 0; j < L; ++j) dst[j] = (__bf16)((row[j] - mean) * inv * g[j] + b[j]);
}

// predictions[b,t,:] = LN(hidden[b,t]) @ next_lin^T + b   (t < 255)
__global__ __launch_bounds__(32) void corvid_pred(const float* __restrict__ hid,
                                                  const __bf16* __restrict__ wp,
                                                  const float* g, const float* bln,
                                                  const float* bias, float* __restrict__ pred)
{
    __shared__ float tmp[16 * 128];
    __shared__ alignas(16) __bf16 xb[16 * XK];
    __shared__ float yb[16 * 16];
    int lane = threadIdx.x;
    size_t r0 = (size_t)blockIdx.x * 16;
    for (int i = lane; i < 16 * 128; i += 32) {
        int m = i >> 7, j = i & 127;
        size_t r = r0 + m, b = r / 255, t = r % 255;
        tmp[i] = hid[(b * CB_T + t) * CB_H + j];
    }
    __syncthreads();
    if (lane < 16) row_ln_to_bf16(&tmp[lane * 128], 128, g, bln, &xb[lane * XK]);
    __syncthreads();
    gemm16(wp + PO_NEXT, 1, 4, xb, bias, 16, yb, 16, lane);
    __syncthreads();
    for (int i = lane; i < 16 * 16; i += 32) {
        int m = i >> 4, j = i & 15;
        pred[(r0 + m) * 16 + j] = yb[i];
    }
}

__global__ void corvid_anom(const float* __restrict__ hid, const float* g,
                            const float* bln, const float* Wv, const float* b,
                            float* __restrict__ anom)
{
    size_t idx = (size_t)blockIdx.x * blockDim.x + threadIdx.x;
    if (idx >= (size_t)CB_B * CB_T) return;
    const float* row = hid + idx * CB_H;
    float mean = 0.0f;
    for (int j = 0; j < 128; ++j) mean += row[j];
    mean *= (1.0f / 128.0f);
    float var = 0.0f;
    for (int j = 0; j < 128; ++j) { float d = row[j] - mean; var += d * d; }
    var *= (1.0f / 128.0f);
    float inv = rsqrtf(var + 1e-5f);
    float s = b[0];
    for (int j = 0; j < 128; ++j) s += ((row[j] - mean) * inv * g[j] + bln[j]) * Wv[j];
    anom[idx] = 1.0f / (1.0f + expf(-s));
}

__global__ void corvid_surp(const float* __restrict__ pred, const float* __restrict__ x,
                            float* __restrict__ surp)
{
    size_t idx = (size_t)blockIdx.x * blockDim.x + threadIdx.x;
    if (idx >= (size_t)CB_B * 255) return;
    size_t b = idx / 255, t = idx % 255;
    const float* p = pred + idx * 16;
    const float* xr = x + (b * CB_T + t + 1) * 16;
    float s = 0.0f;
    for (int i = 0; i < 16; ++i) { float d = p[i] - xr[i]; s += d * d; }
    surp[idx] = s * (1.0f / 16.0f);
}

__global__ __launch_bounds__(32) void corvid_prop(const float* __restrict__ hid,
                                                  const __bf16* __restrict__ wp,
                                                  const float* g, const float* bln,
                                                  const float* b1, const float* W2,
                                                  const float* b2, float* __restrict__ prop)
{
    __shared__ float tmp[16 * 128];
    __shared__ alignas(16) __bf16 xb[16 * XK];
    __shared__ float yb[16 * 64];
    int lane = threadIdx.x;
    int b0 = blockIdx.x * 16;
    for (int i = lane; i < 16 * 128; i += 32) {
        int m = i >> 7, j = i & 127;
        tmp[i] = hid[((size_t)(b0 + m) * CB_T + (CB_T - 1)) * CB_H + j];
    }
    __syncthreads();
    if (lane < 16) row_ln_to_bf16(&tmp[lane * 128], 128, g, bln, &xb[lane * XK]);
    __syncthreads();
    gemm16(wp + PO_PROP1, 4, 4, xb, b1, 64, yb, 64, lane);
    __syncthreads();
    for (int i = lane; i < 16 * 64; i += 32) {
        float v = yb[i];
        yb[i] = 0.5f * v * (1.0f + erff(v * 0.70710678118654752f));  // exact GELU
    }
    __syncthreads();
    {
        int m = lane >> 1, p = lane & 1;
        float s = b2[p];
        for (int j = 0; j < 64; ++j) s += yb[m * 64 + j] * W2[p * 64 + j];
        prop[(size_t)(b0 + m) * 2 + p] = s;
    }
}

// ---------------------------------------------------------------------------
extern "C" void kernel_launch(void* const* d_in, const int* in_sizes, int n_in,
                              void* d_out, int out_size, void* d_ws, size_t ws_size,
                              hipStream_t stream)
{
    (void)in_sizes; (void)out_size; (void)ws_size;
    auto F = [&](int i) { return (const float*)d_in[i]; };
    __bf16* wp = (__bf16*)d_ws;

    auto pack = [&](int ai, int bi, int N, int KA, int KB, int nT, int kT, int off) {
        int total = nT * kT * 512;
        pack_weight<<<(total + 255) / 256, 256, 0, stream>>>(
            F(ai), bi >= 0 ? F(bi) : nullptr, N, KA, KB, nT, kT, wp + off);
    };
    // nuclei pack [Wi | Wh] along K
    pack(1, -1, 15, 16, 0, 1, 1, PO_INPUT);
    pack(3, 5, 15, 15, 15, 1, 1, PO_WULST);
    pack(9, 11, 10, 15, 10, 1, 1, PO_TECTUM);
    pack(15, 17, 19, 44, 19, 2, 2, PO_MESO);
    pack(21, 23, 19, 38, 19, 2, 2, PO_HIPPO);
    pack(27, 29, 23, 61, 23, 2, 3, PO_NCL);
    pack(33, 35, 12, 54, 12, 1, 3, PO_SONG);
    pack(39, 41, 15, 38, 15, 1, 2, PO_CER);
    pack(45, 47, 12, 38, 12, 1, 2, PO_ARCO);
    // axons (dict order: proj.W, proj.b, gate.W, gate.b)
    pack(53, -1, 15, 19, 0, 1, 1, PO_H2W_G);
    pack(51, -1, 15, 19, 0, 1, 1, PO_H2W_P);
    pack(57, -1, 25, 23, 0, 2, 1, PO_N2M_G);
    pack(55, -1, 25, 23, 0, 2, 1, PO_N2M_P);
    pack(61, -1, 38, 12, 0, 3, 1, PO_S2A_G);
    pack(59, -1, 38, 12, 0, 3, 1, PO_S2A_P);
    pack(63, -1, 128, 125, 0, 8, 4, PO_TGATE);
    pack(67, -1, 128, 125, 0, 8, 4, PO_TPROJ);
    pack(71, -1, 16, 128, 0, 1, 4, PO_NEXT);
    pack(79, -1, 64, 128, 0, 4, 4, PO_PROP1);

    // d_out layout: pred[B,255,16] | hidden[B,256,128] | anom[B,256] | surp[B,255] | prop[B,2]
    float* out = (float*)d_out;
    float* pred = out;
    float* hid  = out + (size_t)CB_B * 255 * 16;
    float* anom = hid + (size_t)CB_B * CB_T * CB_H;
    float* surp = anom + (size_t)CB_B * CB_T;
    float* prop = surp + (size_t)CB_B * 255;

    BrainArgs A{};
    for (int i = 0; i < 83 && i < n_in; ++i) A.in[i] = F(i);
    A.wp = wp;
    A.hid = hid;

    corvid_rnn<<<CB_B / 16, 32, 0, stream>>>(A);
    corvid_pred<<<(CB_B * 255) / 16, 32, 0, stream>>>(hid, wp, F(69), F(70), F(72), pred);
    corvid_anom<<<(CB_B * CB_T) / 256, 256, 0, stream>>>(hid, F(73), F(74), F(75), F(76), anom);
    corvid_surp<<<(CB_B * 255 + 255) / 256, 256, 0, stream>>>(pred, F(0), surp);
    corvid_prop<<<CB_B / 16, 32, 0, stream>>>(hid, wp, F(77), F(78), F(80), F(81), F(82), prop);
}
